// CrossAttention_39444979646643
// MI455X (gfx1250) — matrix operand order repaired and checked by reading, not compiled, not run on hardware
//
#include <hip/hip_runtime.h>
#include <hip/hip_bf16.h>

// CDNA5 (gfx1250) cross-attention:
//   3x [16384x1024 @ 1024x1024] projections: f16 WMMA, f32 accum,
//     weight tiles staged into LDS via the Tensor Data Mover (TDM)
//   16384 batched 32x32 attention blocks: f16 WMMA + in-register softmax

typedef __attribute__((ext_vector_type(16))) _Float16 v16h;
typedef __attribute__((ext_vector_type(8)))  float    v8f;

// TDM descriptor operand types (this toolchain: 6-arg builtin,
//   (v4u g0, v8i g1, v4i g2, v4i g3, v8i g4, i32 cpol))
typedef unsigned int v4u __attribute__((ext_vector_type(4)));
typedef int          v8i __attribute__((ext_vector_type(8)));
typedef int          v4i __attribute__((ext_vector_type(4)));

#if defined(__has_builtin)
#if __has_builtin(__builtin_amdgcn_tensor_load_to_lds) && \
    __has_builtin(__builtin_amdgcn_s_wait_tensorcnt)
#define USE_TDM 1
#endif
#endif
#ifndef USE_TDM
#define USE_TDM 0
#endif

#define WMMA_F16(A, B, C)                                                     \
  __builtin_amdgcn_wmma_f32_16x16x32_f16(false, (A), false, (B), (short)0,    \
                                         (C), false, false)

// ---------------------------------------------------------------------------
// Fragment helpers (gfx1250 16-bit WMMA layouts, wave32):
//  A (16x32, MxK): lane 0-15 -> M=lane, lane 16-31 -> M=lane-16.
//    halfs 0..7  = K(off..off+7), halfs 8..15 = K(off+16..off+23),
//    off = (lane/16)*8 within the 32-wide K step.
//  B (32x16, KxN): lane -> N=lane%16; halfs 0..7 = K(off..off+7), halfs 8..15
//    = K(off+16..off+23), off = (lane/16)*8, same order as A.
//  C/D (16x16 f32): VGPR r -> M = r + (lane/16)*8, N = lane%16.
// ---------------------------------------------------------------------------

__device__ __forceinline__ v16h load_a_frag_f32(const float* __restrict__ row,
                                                int off) {
  v16h a;
#pragma unroll
  for (int h = 0; h < 8; ++h) {
    a[h]     = (_Float16)row[off + h];
    a[h + 8] = (_Float16)row[off + 16 + h];
  }
  return a;
}

__device__ __forceinline__ v16h lds_b_frag32(const _Float16* __restrict__ lds,
                                             int nrow, int kc0) {
  // B operand from an LDS-resident [rows x 32] f16 tile (K contiguous).
  v16h b;
#pragma unroll
  for (int h = 0; h < 16; ++h) b[h] = lds[nrow * 32 + (kc0 >> 1) + (h < 8 ? h : h + 8)];
  return b;
}

// ---------------------------------------------------------------------------
// Weight pre-conversion: f32 -> f16, row-major (K contiguous), 1M elements.
// ---------------------------------------------------------------------------
__global__ __launch_bounds__(256) void cvt_w_f16(const float* __restrict__ src,
                                                 _Float16* __restrict__ dst) {
  const int idx = (blockIdx.x * 256 + threadIdx.x) * 4;
  const float4 v = *(const float4*)(src + idx);
  dst[idx + 0] = (_Float16)v.x;
  dst[idx + 1] = (_Float16)v.y;
  dst[idx + 2] = (_Float16)v.z;
  dst[idx + 3] = (_Float16)v.w;
}

// ---------------------------------------------------------------------------
// Projection GEMM: Y[m,n] = sum_k X[m,k] * Wh[n,k] + bias[n]   (f16 out)
// M=16384, N=1024, K=1024. Block = 8 waves (2 M x 4 N), wave tile = 32x32.
// Per K-step the block's 128x32 f16 weight tile is DMA'd into LDS by the
// Tensor Data Mover (wave 0 issues, s_wait_tensorcnt + barrier publishes).
// A operand is streamed f32 -> cvt f16 in-register (single pass over X).
// ---------------------------------------------------------------------------
__global__ __launch_bounds__(256) void proj_gemm(
    const float* __restrict__ X, const _Float16* __restrict__ Wh,
    const float* __restrict__ bias, _Float16* __restrict__ Y) {
  __shared__ _Float16 ldsW[128 * 32];  // 8 KB weight tile [n_local][k_local]

  const int lane = threadIdx.x & 31;
  const int wave = threadIdx.x >> 5;
  const int half = lane >> 4;  // 0/1
  const int l15  = lane & 15;

  const int m0      = blockIdx.x * 64 + (wave >> 2) * 32;  // wave M base
  const int nblk    = blockIdx.y * 128;                    // block N base
  const int n0      = nblk + (wave & 3) * 32;              // wave N base
  const int nl_base = (wave & 3) * 32;                     // wave N base in LDS

  const float* xr0 = X + (size_t)(m0 + l15) * 1024;
  const float* xr1 = X + (size_t)(m0 + 16 + l15) * 1024;

  const int aoff = half << 3;  // A-frag K chunk base within a 32-step
  const int kc0  = half << 4;  // B-frag K base within a 32-step

  v8f acc00{}, acc01{}, acc10{}, acc11{};

  for (int k0 = 0; k0 < 1024; k0 += 32) {
    __syncthreads();  // previous tile fully consumed
#if USE_TDM
    if (wave == 0) {
      // Tensor DMA: 2D tile, 32 halfs (dim0) x 128 rows (dim1), row stride
      // 1024 halfs, data_size = 2 bytes, into ldsW.
      const unsigned long long ga =
          (unsigned long long)(uintptr_t)(Wh + (size_t)nblk * 1024 + k0);
      const unsigned ldsoff = (unsigned)(uintptr_t)&ldsW[0];
      v4u g0 = {1u,                                  // count=1 valid descriptor
                ldsoff,                              // lds_addr (bytes)
                (unsigned)(ga & 0xffffffffu),        // global_addr[31:0]
                (unsigned)((ga >> 32) & 0x01ffffffu) // global_addr[56:32]
                    | (2u << 30)};                   // type=2 ("image")
      v8i g1 = {(int)(1u << 16),     // workgroup_mask=0, data_size=1 (2 bytes)
                (int)(1024u << 16),  // tensor_dim0 = 1024 (lo16)
                (int)(1024u << 16),  // tensor_dim0 hi=0 | tensor_dim1 = 1024
                (int)(32u << 16),    // tensor_dim1 hi=0 | tile_dim0 = 32
                (int)128,            // tile_dim1 = 128, tile_dim2 = 0
                (int)1024,           // tensor_dim0_stride[31:0] = 1024
                0, 0};               // stride hi, tensor_dim1_stride = 0
      v4i g2 = {0, 0, 0, 0};
      v4i g3 = {0, 0, 0, 0};
      v8i g4 = {0, 0, 0, 0, 0, 0, 0, 0};
      __builtin_amdgcn_tensor_load_to_lds(g0, g1, g2, g3, g4, 0);
      __builtin_amdgcn_s_wait_tensorcnt(0);
    }
#else
    // Fallback: cooperative 16B-chunk copy of the 128x32 tile.
    for (int c = threadIdx.x; c < 128 * 4; c += 256) {
      const int r = c >> 2, q = c & 3;
      const float4* src =
          (const float4*)(Wh + (size_t)(nblk + r) * 1024 + k0 + q * 8);
      *(float4*)&ldsW[r * 32 + q * 8] = *src;
    }
#endif
    __syncthreads();  // tile visible to all waves

    __builtin_prefetch(xr0 + k0 + 128, 0, 1);  // global_prefetch_b8
    __builtin_prefetch(xr1 + k0 + 128, 0, 1);

    v16h a0 = load_a_frag_f32(xr0, k0 + aoff);
    v16h a1 = load_a_frag_f32(xr1, k0 + aoff);
    v16h b0 = lds_b_frag32(ldsW, nl_base + l15, kc0);
    v16h b1 = lds_b_frag32(ldsW, nl_base + 16 + l15, kc0);

    acc00 = WMMA_F16(a0, b0, acc00);
    acc01 = WMMA_F16(a0, b1, acc01);
    acc10 = WMMA_F16(a1, b0, acc10);
    acc11 = WMMA_F16(a1, b1, acc11);
  }

  const float bn0 = bias[n0 + l15];
  const float bn1 = bias[n0 + 16 + l15];
#pragma unroll
  for (int r = 0; r < 8; ++r) {
    const int mg0 = m0 + r + (half << 3);
    const int mg1 = m0 + 16 + r + (half << 3);
    Y[(size_t)mg0 * 1024 + n0 + l15]      = (_Float16)(acc00[r] + bn0);
    Y[(size_t)mg0 * 1024 + n0 + 16 + l15] = (_Float16)(acc01[r] + bn1);
    Y[(size_t)mg1 * 1024 + n0 + l15]      = (_Float16)(acc10[r] + bn0);
    Y[(size_t)mg1 * 1024 + n0 + 16 + l15] = (_Float16)(acc11[r] + bn1);
  }
}

// ---------------------------------------------------------------------------
// Attention stage helpers
// ---------------------------------------------------------------------------
__device__ __forceinline__ v16h gather_q_frag(const _Float16* __restrict__ q,
                                              size_t qrow, int chi) {
  // A[m][c] = qh[qbase + c*32768 + m*32]; qrow = qbase + m*32
  v16h a;
#pragma unroll
  for (int h = 0; h < 8; ++h) {
    a[h]     = q[qrow + (size_t)(chi + h) * 32768u];
    a[h + 8] = q[qrow + (size_t)(chi + 16 + h) * 32768u];
  }
  return a;
}

__device__ __forceinline__ v16h gather_k_frag(const _Float16* __restrict__ krow,
                                              int kc0, int n) {
  // B[c][n] = krow[c*32 + n]
  v16h b;
#pragma unroll
  for (int h = 0; h < 16; ++h) b[h] = krow[(size_t)((kc0 >> 1) + (h < 8 ? h : h + 8)) * 32 + n];
  return b;
}

__device__ __forceinline__ v16h load_v_frag(const _Float16* __restrict__ vrow_m,
                                            int chi) {
  // A[m][l] = vrow[m*32 + l] (contiguous per row)
  v16h a;
#pragma unroll
  for (int h = 0; h < 8; ++h) {
    a[h]     = vrow_m[chi + h];
    a[h + 8] = vrow_m[chi + 16 + h];
  }
  return a;
}

// Row softmax directly in the WMMA C/D layout: each 32-wide row lives in one
// VGPR index r across a 16-lane half, split over the two N-tiles (e0, e1).
__device__ __forceinline__ void softmax_rows(v8f& e0, v8f& e1) {
#pragma unroll
  for (int r = 0; r < 8; ++r) {
    float mx = fmaxf(e0[r], e1[r]);
#pragma unroll
    for (int s = 8; s >= 1; s >>= 1) mx = fmaxf(mx, __shfl_xor(mx, s, 32));
    const float p0 = __expf(e0[r] - mx);
    const float p1 = __expf(e1[r] - mx);
    float sum = p0 + p1;
#pragma unroll
    for (int s = 8; s >= 1; s >>= 1) sum += __shfl_xor(sum, s, 32);
    const float inv = 1.0f / sum;
    e0[r] = p0 * inv;
    e1[r] = p1 * inv;
  }
}

// ---------------------------------------------------------------------------
// Batched 32x32 attention: one wave per (b,i,j) batch, 8 waves per block.
//   E = Qm @ Km ; A = softmax(E) ; O = Vm @ A^T ; transposed scatter to out.
// ---------------------------------------------------------------------------
__global__ __launch_bounds__(256) void attn_kernel(
    const _Float16* __restrict__ qh, const _Float16* __restrict__ kh,
    const _Float16* __restrict__ vh, float* __restrict__ out) {
  __shared__ _Float16 att_lds[8][32 * 32];

  const int lane  = threadIdx.x & 31;
  const int wave  = threadIdx.x >> 5;
  const int batch = blockIdx.x * 8 + wave;
  const int j = batch & 31;
  const int i = (batch >> 5) & 31;
  const int b = batch >> 10;

  const int half = lane >> 4;
  const int l15  = lane & 15;
  const int chi  = half << 3;  // A-frag K chunk base
  const int kc0  = half << 4;  // B-frag K base

  const size_t qbase = ((size_t)b << 20) + (size_t)i * 1024 + j;
  const size_t rkv   = ((size_t)b * 1024 + (size_t)i * 32 + j) << 10;
  const _Float16* krow = kh + rkv;
  const _Float16* vrow = vh + rkv;

  // ---- energy = Qm @ Km (4 WMMAs, f32 accum) ----
  v16h aq0 = gather_q_frag(qh, qbase + (size_t)(l15)*32, chi);
  v16h aq1 = gather_q_frag(qh, qbase + (size_t)(16 + l15) * 32, chi);
  v16h bk0 = gather_k_frag(krow, kc0, l15);
  v16h bk1 = gather_k_frag(krow, kc0, 16 + l15);

  v8f z{};
  v8f e00 = WMMA_F16(aq0, bk0, z);
  v8f e01 = WMMA_F16(aq0, bk1, z);
  v8f e10 = WMMA_F16(aq1, bk0, z);
  v8f e11 = WMMA_F16(aq1, bk1, z);

  // ---- softmax over the 32 columns of each row (register-resident) ----
  softmax_rows(e00, e01);
  softmax_rows(e10, e11);

  // ---- re-layout attention as a B operand via per-wave LDS slice ----
  _Float16* lds = att_lds[wave];
#pragma unroll
  for (int r = 0; r < 8; ++r) {
    const int l0 = r + (half << 3);
    const int l1 = 16 + r + (half << 3);
    lds[l0 * 32 + l15]      = (_Float16)e00[r];
    lds[l0 * 32 + 16 + l15] = (_Float16)e01[r];
    lds[l1 * 32 + l15]      = (_Float16)e10[r];
    lds[l1 * 32 + 16 + l15] = (_Float16)e11[r];
  }
  __syncthreads();

  v16h ba0 = lds_b_frag32(lds, l15, kc0);
  v16h ba1 = lds_b_frag32(lds, 16 + l15, kc0);
  v16h av0 = load_v_frag(vrow + (size_t)(l15)*32, chi);
  v16h av1 = load_v_frag(vrow + (size_t)(16 + l15) * 32, chi);

  // ---- O = Vm @ A^T (4 WMMAs) ----
  v8f o00 = WMMA_F16(av0, ba0, z);
  v8f o01 = WMMA_F16(av0, ba1, z);
  v8f o10 = WMMA_F16(av1, ba0, z);
  v8f o11 = WMMA_F16(av1, ba1, z);

  // ---- transposed scatter: out[((b*32+j)*32+n)*1024 + m*32 + i] ----
  float* ob = out + ((((size_t)b * 32 + j) * 32) << 10) + i;
#pragma unroll
  for (int r = 0; r < 8; ++r) {
    const int m0r = r + (half << 3);
    const int m1r = 16 + r + (half << 3);
    ob[(size_t)(l15)*1024 + m0r * 32]        = o00[r];
    ob[(size_t)(16 + l15) * 1024 + m0r * 32] = o01[r];
    ob[(size_t)(l15)*1024 + m1r * 32]        = o10[r];
    ob[(size_t)(16 + l15) * 1024 + m1r * 32] = o11[r];
  }
}

// ---------------------------------------------------------------------------
extern "C" void kernel_launch(void* const* d_in, const int* in_sizes, int n_in,
                              void* d_out, int out_size, void* d_ws,
                              size_t ws_size, hipStream_t stream) {
  (void)in_sizes; (void)n_in; (void)out_size; (void)ws_size;

  const float* x1 = (const float*)d_in[0];
  const float* x2 = (const float*)d_in[1];
  const float* Wq = (const float*)d_in[2];
  const float* bq = (const float*)d_in[3];
  const float* Wk = (const float*)d_in[4];
  const float* bk = (const float*)d_in[5];
  const float* Wv = (const float*)d_in[6];
  const float* bv = (const float*)d_in[7];
  float* out = (float*)d_out;

  const size_t MN = (size_t)16384 * 1024;  // 16.7M
  const size_t WN = (size_t)1024 * 1024;   // 1M
  _Float16* qh  = (_Float16*)d_ws;
  _Float16* kh  = qh + MN;
  _Float16* vh  = kh + MN;
  _Float16* wqh = vh + MN;
  _Float16* wkh = wqh + WN;
  _Float16* wvh = wkh + WN;  // total f16 workspace: 102 MB

  // Weights f32 -> f16 (one-time, 2 MB each).
  cvt_w_f16<<<WN / (256 * 4), 256, 0, stream>>>(Wq, wqh);
  cvt_w_f16<<<WN / (256 * 4), 256, 0, stream>>>(Wk, wkh);
  cvt_w_f16<<<WN / (256 * 4), 256, 0, stream>>>(Wv, wvh);

  dim3 ggrid(16384 / 64, 1024 / 128);  // 256 x 8 blocks, 8 waves each
  proj_gemm<<<ggrid, 256, 0, stream>>>(x1, wqh, bq, qh);
  proj_gemm<<<ggrid, 256, 0, stream>>>(x2, wkh, bk, kh);
  proj_gemm<<<ggrid, 256, 0, stream>>>(x2, wvh, bv, vh);

  attn_kernel<<<16384 / 8, 256, 0, stream>>>(qh, kh, vh, out);
}
